// AudioSNN_9706626088992
// MI455X (gfx1250) — compile-verified, matrix operand order checked by reading, not run on hardware
//
#include <hip/hip_runtime.h>
#include <hip/hip_bf16.h>

typedef __attribute__((ext_vector_type(16))) _Float16 v16h;
typedef __attribute__((ext_vector_type(8)))  _Float16 v8h;
typedef __attribute__((ext_vector_type(8)))  float    v8f;

#define BETA 0.95f
#define THR  1.0f

// ---------------------------------------------------------------------------
// Fragment helpers (CDNA5 WMMA 16x16x32 f16 layout):
// A (16xK) / B (Kx16, column n per lane): lane L -> row/col = L&15.
// halves 0..7  = K[kb + (L<16?0:8) .. +7]
// halves 8..15 = K[kb + 16 + (L<16?0:8) .. +7]
// -> two contiguous 8-half runs => two b128 loads.
// ---------------------------------------------------------------------------
__device__ __forceinline__ v16h cat16(v8h lo, v8h hi) {
    v16h r;
#pragma unroll
    for (int i = 0; i < 8; ++i) { r[i] = lo[i]; r[i + 8] = hi[i]; }
    return r;
}

__device__ __forceinline__ v16h load_frag(const _Float16* base, int strideHalves,
                                          int row, int kb, int lane) {
    const _Float16* p = base + (size_t)row * strideHalves + kb + ((lane & 16) ? 8 : 0);
    v8h lo = *(const v8h*)p;
    v8h hi = *(const v8h*)(p + 16);
    return cat16(lo, hi);
}

__device__ __forceinline__ v8f wmma_f16(v16h a, v16h b, v8f c) {
    return __builtin_amdgcn_wmma_f32_16x16x32_f16(false, a, false, b, (short)0, c,
                                                  false, false);
}

// ---------------------------------------------------------------------------
// Weight conversion kernels (f32 -> f16 once per call; cheap, bandwidth-bound)
// ---------------------------------------------------------------------------
__global__ void cvt_f32_f16(const float* __restrict__ src,
                            _Float16* __restrict__ dst, int n) {
    int i = blockIdx.x * blockDim.x + threadIdx.x;
    if (i < n) dst[i] = (_Float16)src[i];
}

// conv2_w [64][32][3][3] f32 -> [64][3*3][32] f16  (K reordered to (kh,kw,ic))
__global__ void cvt_w2_reorder(const float* __restrict__ src, _Float16* __restrict__ dst) {
    int i = blockIdx.x * blockDim.x + threadIdx.x;
    if (i >= 64 * 288) return;
    int n = i / 288, r = i % 288;
    int f = r >> 5, ic = r & 31;          // f = kh*3+kw
    int kh = f / 3, kw = f % 3;
    dst[i] = (_Float16)src[((n * 32 + ic) * 3 + kh) * 3 + kw];
}

// fc3_w [10,128] f32 -> padded [16,128] f16 (rows 10..15 = 0)
__global__ void cvt_fc3_pad(const float* __restrict__ src, _Float16* __restrict__ dst) {
    int i = blockIdx.x * blockDim.x + threadIdx.x;
    if (i >= 16 * 128) return;
    dst[i] = (i < 10 * 128) ? (_Float16)src[i] : (_Float16)0.f;
}

// ---------------------------------------------------------------------------
// Kernel 1: fused conv1+spike+pool  ->  conv2 (implicit-GEMM WMMA) +spike+pool
// One workgroup (256 thr = 8 waves) per batch image.
//   conv2 GEMM view: M = 16 output columns (tile = one conv row h),
//                    N = 64 out channels, K = 9 taps x 32 in-channels = 288.
//   K ordering (kh,kw,ic) + channel-minor halo-padded c1 [34][18][32] makes
//   every A-fragment two contiguous ds_load_b128 (no gathers, no spills).
//   Wave w handles pooled rows ph in {2w, 2w+1}; each ph = conv rows
//   {2ph, 2ph+1} = two m-tiles accumulated together so the 2x2 pool
//   happens entirely inside the D fragments.
// ---------------------------------------------------------------------------
__launch_bounds__(256)
__global__ void conv_pipeline(const float* __restrict__ x,
                              const float* __restrict__ w1,
                              const float* __restrict__ b1,
                              const _Float16* __restrict__ w2r, // [64][9][32] f16
                              const float* __restrict__ b2,
                              _Float16* __restrict__ xflat) {   // [1024][8192] f16
    __shared__ __align__(16) float     xsp[66][34];          // 8.8 KB, zero halo
    __shared__ __align__(16) _Float16  c1v[34 * 18 * 32];    // 38.3 KB, ch-minor, zero halo
    __shared__ float w1s[32 * 9];
    __shared__ float b1s[32];

    const int b    = blockIdx.x;
    const int tid  = threadIdx.x;
    const int lane = tid & 31;
    const int wave = tid >> 5;

    // zero halos + load input slab + conv1 weights
    for (int i = tid; i < 66 * 34; i += 256) ((float*)xsp)[i] = 0.f;
    for (int i = tid; i < 34 * 18 * 32; i += 256) c1v[i] = (_Float16)0.f;
    for (int i = tid; i < 32 * 9; i += 256) w1s[i] = w1[i];
    if (tid < 32) b1s[tid] = b1[tid];
    __syncthreads();

    const float* xb = x + (size_t)b * 64 * 32;
    for (int i = tid; i < 64 * 32; i += 256) xsp[(i >> 5) + 1][(i & 31) + 1] = xb[i];
    __syncthreads();

    // ---- conv1 + spike + avgpool2 -> c1v interior, values in {0,.25,.5,.75,1}
    for (int i = tid; i < 32 * 32 * 16; i += 256) {
        int pw = i & 15, ph = (i >> 4) & 31, c = i >> 9;
        float s = 0.f;
#pragma unroll
        for (int dh = 0; dh < 2; ++dh)
#pragma unroll
            for (int dw = 0; dw < 2; ++dw) {
                int y = 2 * ph + dh, xq = 2 * pw + dw;
                float acc = b1s[c];
#pragma unroll
                for (int kh = 0; kh < 3; ++kh)
#pragma unroll
                    for (int kw = 0; kw < 3; ++kw)
                        acc += w1s[c * 9 + kh * 3 + kw] * xsp[y + kh][xq + kw];
                s += (acc > THR) ? 1.f : 0.f;
            }
        // channel-minor: [y=ph+1][x=pw+1][c]
        c1v[((ph + 1) * 18 + (pw + 1)) * 32 + c] = (_Float16)(0.25f * s);
    }
    __syncthreads();

    // ---- conv2 as implicit GEMM with WMMA, fused spike + pool -> xflat
    const int wx   = lane & 15;              // output column in tile
    const int koff = (lane & 16) ? 8 : 0;    // lane-half K sub-offset

    for (int q = 0; q < 2; ++q) {
        const int ph = 2 * wave + q;         // pooled output row 0..15
        const int h0 = 2 * ph;               // conv rows h0, h0+1
        // base of gather for (y = h0+kh padded, x = wx+kw padded, ic = koff)
        const int gb = (h0 * 18 + wx) * 32 + koff;
        v8f acc0[4] = {}, acc1[4] = {};

#pragma unroll
        for (int f = 0; f < 9; ++f) {        // one (kh,kw) tap = one K-chunk of 32
            const int kh = f / 3, kw = f % 3;
            const _Float16* p0 = &c1v[gb + (kh * 18 + kw) * 32];
            v16h a0 = cat16(*(const v8h*)p0, *(const v8h*)(p0 + 16));
            const _Float16* p1 = p0 + 18 * 32;   // next conv row (h0+1)
            v16h a1 = cat16(*(const v8h*)p1, *(const v8h*)(p1 + 16));
#pragma unroll
            for (int nt = 0; nt < 4; ++nt) { // 64 out channels = 4 N-tiles
                v16h bf = load_frag(w2r, 288, nt * 16 + wx, 32 * f, lane);
                acc0[nt] = wmma_f16(a0, bf, acc0[nt]);
                acc1[nt] = wmma_f16(a1, bf, acc1[nt]);
            }
        }

        // spike + 2x2 pool inside D fragments:
        //   D row r of lane L is conv column w = r + 8*(L>>4);
        //   w-pairs (2rr,2rr+1) are regs (2rr,2rr+1); h-pair is acc0/acc1.
#pragma unroll
        for (int nt = 0; nt < 4; ++nt) {
            int n = nt * 16 + wx;
            float bias = b2[n];
#pragma unroll
            for (int rr = 0; rr < 4; ++rr) {
                float s = ((acc0[nt][2 * rr]     + bias > THR) ? 1.f : 0.f)
                        + ((acc0[nt][2 * rr + 1] + bias > THR) ? 1.f : 0.f)
                        + ((acc1[nt][2 * rr]     + bias > THR) ? 1.f : 0.f)
                        + ((acc1[nt][2 * rr + 1] + bias > THR) ? 1.f : 0.f);
                int pw = rr + 4 * (lane >> 4);
                xflat[(size_t)b * 8192 + n * 128 + ph * 8 + pw] = (_Float16)(0.25f * s);
            }
        }
    }
}

// ---------------------------------------------------------------------------
// Kernel 2: cur3 = xflat[1024,8192] @ fc1_w[256,8192]^T + b  (f16 WMMA, f32 acc)
// One wave per 16x16 output tile; K = 8192 = 256 WMMA steps (unroll 4 so the
// compiler clauses 16 b128 loads per 4 WMMAs).
// ---------------------------------------------------------------------------
__launch_bounds__(256)
__global__ void fc1_gemm(const _Float16* __restrict__ A,   // [1024][8192]
                         const _Float16* __restrict__ B,   // [256][8192]
                         const float* __restrict__ bias,
                         float* __restrict__ C) {          // [1024][256] f32
    const int lane = threadIdx.x & 31, wave = threadIdx.x >> 5;
    const int tile = blockIdx.x * 8 + wave;   // 0..1023
    const int m0 = (tile >> 4) * 16, n0 = (tile & 15) * 16;
    const int ar = m0 + (lane & 15), br = n0 + (lane & 15);

    v8f acc = {};
#pragma unroll 4
    for (int kb = 0; kb < 8192; kb += 32) {
        v16h a  = load_frag(A, 8192, ar, kb, lane);
        v16h bf = load_frag(B, 8192, br, kb, lane);
        acc = wmma_f16(a, bf, acc);
    }
    const int n = n0 + (lane & 15);
    const float bv = bias[n];
#pragma unroll
    for (int r = 0; r < 8; ++r) {
        int m = m0 + r + 8 * (lane >> 4);
        C[(size_t)m * 256 + n] = acc[r] + bv;
    }
}

// ---------------------------------------------------------------------------
// Kernel 3: 25-step LIF recurrence. cur4[b,:] depends only on spk3[b,:], so
// each workgroup owns 16 batch rows and runs ALL timesteps with LDS-resident
// membrane state — no inter-workgroup communication, WMMA for fc2/fc3.
// cur3 / biases are hoisted into registers before the time loop.
// ---------------------------------------------------------------------------
__launch_bounds__(256)
__global__ void snn_steps(const float* __restrict__ cur3,      // [1024][256]
                          const _Float16* __restrict__ w2,     // [128][256] f16
                          const float* __restrict__ b2,
                          const _Float16* __restrict__ w3,     // [16][128] f16 padded
                          const float* __restrict__ b3,
                          float* __restrict__ out) {           // [25][1024][10]
    __shared__ __align__(16) float     mem3[16][256];  // 16 KB
    __shared__ __align__(16) float     mem4[16][128];  //  8 KB
    __shared__               float     mem5[16][10];
    __shared__ __align__(16) _Float16  spk3[16][256];  //  8 KB
    __shared__ __align__(16) _Float16  spk4[16][128];  //  4 KB
    __shared__ __align__(16) float     cur4[16][128];  //  8 KB
    __shared__ __align__(16) float     cur5[16][16];   //  1 KB

    const int g    = blockIdx.x;          // batch rows [16g, 16g+16)
    const int tid  = threadIdx.x;
    const int lane = tid & 31, wave = tid >> 5;
    const int koff = (lane & 16) ? 8 : 0;

    // hoist per-thread constants: input currents and biases
    float c3r[16];
#pragma unroll
    for (int j = 0; j < 16; ++j)
        c3r[j] = cur3[(size_t)(16 * g + j) * 256 + tid];
    float b2r[8];
#pragma unroll
    for (int j = 0; j < 8; ++j) b2r[j] = b2[(j * 256 + tid) & 127];
    const float b3r = (tid < 160) ? b3[tid % 10] : 0.f;

    for (int j = 0; j < 16; ++j) mem3[j][tid] = 0.f;
    for (int j = 0; j < 8; ++j) { int e = j * 256 + tid; mem4[e >> 7][e & 127] = 0.f; }
    if (tid < 160) mem5[tid / 10][tid % 10] = 0.f;
    __syncthreads();

    for (int step = 0; step < 25; ++step) {
        // ---- layer 3 LIF (subtract reset based on PREVIOUS mem)
#pragma unroll
        for (int j = 0; j < 16; ++j) {
            float m = mem3[j][tid];
            float r = (m > THR) ? THR : 0.f;
            m = BETA * m + c3r[j] - r;
            mem3[j][tid] = m;
            spk3[j][tid] = (m > THR) ? (_Float16)1.f : (_Float16)0.f;
        }
        __syncthreads();

        // ---- fc2: 16x128 = 8 N-tiles, wave w owns tile w; K = 256 = 8 steps
        {
            v8f acc = {};
            const int br = wave * 16 + (lane & 15);
#pragma unroll
            for (int kb = 0; kb < 256; kb += 32) {
                const _Float16* ap = &spk3[lane & 15][kb + koff];
                v16h a  = cat16(*(const v8h*)ap, *(const v8h*)(ap + 16));
                v16h bf = load_frag(w2, 256, br, kb, lane);
                acc = wmma_f16(a, bf, acc);
            }
#pragma unroll
            for (int r = 0; r < 8; ++r)
                cur4[r + 8 * (lane >> 4)][wave * 16 + (lane & 15)] = acc[r];
        }
        __syncthreads();

        // ---- layer 4 LIF
#pragma unroll
        for (int j = 0; j < 8; ++j) {
            int e = j * 256 + tid, mi = e >> 7, ni = e & 127;
            float m = mem4[mi][ni];
            float r = (m > THR) ? THR : 0.f;
            m = BETA * m + (cur4[mi][ni] + b2r[j]) - r;
            mem4[mi][ni] = m;
            spk4[mi][ni] = (m > THR) ? (_Float16)1.f : (_Float16)0.f;
        }
        __syncthreads();

        // ---- fc3: single 16x16 tile (N padded 10->16), K = 128 = 4 steps
        if (wave == 0) {
            v8f acc = {};
#pragma unroll
            for (int kb = 0; kb < 128; kb += 32) {
                const _Float16* ap = &spk4[lane & 15][kb + koff];
                v16h a  = cat16(*(const v8h*)ap, *(const v8h*)(ap + 16));
                v16h bf = load_frag(w3, 128, lane & 15, kb, lane);
                acc = wmma_f16(a, bf, acc);
            }
#pragma unroll
            for (int r = 0; r < 8; ++r)
                cur5[r + 8 * (lane >> 4)][lane & 15] = acc[r];
        }
        __syncthreads();

        // ---- layer 5 LIF + record output spikes
        if (tid < 160) {
            int mi = tid / 10, ni = tid % 10;
            float m = mem5[mi][ni];
            float r = (m > THR) ? THR : 0.f;
            m = BETA * m + (cur5[mi][ni] + b3r) - r;
            mem5[mi][ni] = m;
            out[(size_t)step * 10240 + (16 * g + mi) * 10 + ni] = (m > THR) ? 1.f : 0.f;
        }
        __syncthreads();
    }
}

// ---------------------------------------------------------------------------
extern "C" void kernel_launch(void* const* d_in, const int* in_sizes, int n_in,
                              void* d_out, int out_size, void* d_ws, size_t ws_size,
                              hipStream_t stream) {
    (void)in_sizes; (void)n_in; (void)out_size; (void)ws_size;

    const float* x   = (const float*)d_in[0];
    const float* c1w = (const float*)d_in[1];
    const float* c1b = (const float*)d_in[2];
    const float* c2w = (const float*)d_in[3];
    const float* c2b = (const float*)d_in[4];
    const float* f1w = (const float*)d_in[5];
    const float* f1b = (const float*)d_in[6];
    const float* f2w = (const float*)d_in[7];
    const float* f2b = (const float*)d_in[8];
    const float* f3w = (const float*)d_in[9];
    const float* f3b = (const float*)d_in[10];
    float* out = (float*)d_out;

    char* ws = (char*)d_ws;
    size_t off = 0;
    auto alloc = [&](size_t bytes) {
        off = (off + 255) & ~(size_t)255;
        char* p = ws + off;
        off += bytes;
        return p;
    };
    _Float16* xflat = (_Float16*)alloc((size_t)1024 * 8192 * 2);  // 16.8 MB
    _Float16* f1w16 = (_Float16*)alloc((size_t)256 * 8192 * 2);   //  4.2 MB
    _Float16* f2w16 = (_Float16*)alloc((size_t)128 * 256 * 2);
    _Float16* f3w16 = (_Float16*)alloc((size_t)16 * 128 * 2);
    _Float16* w2r   = (_Float16*)alloc((size_t)64 * 288 * 2);
    float*    cur3  = (float*)   alloc((size_t)1024 * 256 * 4);   //  1.0 MB

    cvt_w2_reorder<<<(64 * 288 + 255) / 256, 256, 0, stream>>>(c2w, w2r);
    cvt_f32_f16<<<(256 * 8192 + 255) / 256, 256, 0, stream>>>(f1w, f1w16, 256 * 8192);
    cvt_f32_f16<<<(128 * 256 + 255) / 256, 256, 0, stream>>>(f2w, f2w16, 128 * 256);
    cvt_fc3_pad<<<8, 256, 0, stream>>>(f3w, f3w16);

    conv_pipeline<<<1024, 256, 0, stream>>>(x, c1w, c1b, w2r, c2b, xflat);
    fc1_gemm<<<128, 256, 0, stream>>>(xflat, f1w16, f1b, cur3);
    snn_steps<<<64, 256, 0, stream>>>(cur3, f2w16, f2b, f3w16, f3b, out);
}
